// DyMNBlock_22771916604133
// MI455X (gfx1250) — compile-verified, matrix operand order checked by reading, not run on hardware
//
#include <hip/hip_runtime.h>

typedef __attribute__((ext_vector_type(16))) _Float16 v16h;
typedef __attribute__((ext_vector_type(8)))  _Float16 v8h;
typedef __attribute__((ext_vector_type(8)))  float    v8f;

#define BB   16
#define CC   64
#define EE   256
#define OO   64
#define CDIM 64
#define KKN  4
#define FD   128
#define TD   250
#define LL   378      // FD + TD
#define FT   32000    // FD * TD
#define EPSF 1e-5f

// ---------------------------------------------------------------- pooling
__global__ void k_pool(const float* __restrict__ x,
                       float* __restrict__ xf, float* __restrict__ xt) {
  __shared__ float xf_l[FD];
  __shared__ float xt_l[TD];
  int row = blockIdx.x;                       // b*C + c
  int tid = threadIdx.x;
  for (int i = tid; i < FD; i += 256) xf_l[i] = 0.f;
  for (int i = tid; i < TD; i += 256) xt_l[i] = 0.f;
  __syncthreads();
  const float* xr = x + (size_t)row * FT;
  for (int i = tid; i < FT; i += 256) {
    float v = xr[i];
    int f = i / TD, t = i - f * TD;
    atomicAdd(&xf_l[f], v);
    atomicAdd(&xt_l[t], v);
  }
  __syncthreads();
  for (int i = tid; i < FD; i += 256) xf[(size_t)row * FD + i] = xf_l[i] * (1.f / TD);
  for (int i = tid; i < TD; i += 256) xt[(size_t)row * TD + i] = xt_l[i] * (1.f / FD);
}

// ------------------------------------------------- joint = jc_w @ ji, + stats
__global__ void k_joint(const float* __restrict__ xf, const float* __restrict__ xt,
                        const float* __restrict__ jc_w,
                        float* __restrict__ joint, float* jsum, float* jsumsq) {
  __shared__ float s[CDIM], sq[CDIM];
  int tid = threadIdx.x;
  if (tid < CDIM) { s[tid] = 0.f; sq[tid] = 0.f; }
  __syncthreads();
  int idx = blockIdx.x * 256 + tid;           // exact: B*CD*L = 387072
  int b   = idx / (CDIM * LL);
  int rem = idx - b * CDIM * LL;
  int o   = rem / LL;
  int l   = rem - o * LL;
  float acc = 0.f;
  const float* w = jc_w + o * CC;
  if (l < FD) {
    const float* p = xf + ((size_t)b * CC) * FD + l;
    for (int c = 0; c < CC; ++c) acc += w[c] * p[(size_t)c * FD];
  } else {
    const float* p = xt + ((size_t)b * CC) * TD + (l - FD);
    for (int c = 0; c < CC; ++c) acc += w[c] * p[(size_t)c * TD];
  }
  joint[idx] = acc;
  atomicAdd(&s[o], acc);
  atomicAdd(&sq[o], acc * acc);
  __syncthreads();
  if (tid < CDIM) { atomicAdd(&jsum[tid], s[tid]); atomicAdd(&jsumsq[tid], sq[tid]); }
}

// ----------------------------------- BN(joint) + hardswish + context (atomic)
__global__ void k_ctx(const float* __restrict__ joint,
                      const float* jsum, const float* jsumsq,
                      const float* jn_g, const float* jn_b,
                      float* __restrict__ jointn, float* __restrict__ ctx) {
  int idx = blockIdx.x * 256 + threadIdx.x;   // exact grid
  int bo  = idx / LL;
  int o   = bo & (CDIM - 1);
  float inv = 1.f / (float)(BB * LL);
  float m = jsum[o] * inv;
  float v = jsumsq[o] * inv - m * m;
  float r = jn_g[o] * rsqrtf(v + EPSF);
  float val = joint[idx] * r + (jn_b[o] - m * r);
  float hs = val * fminf(fmaxf(val + 3.f, 0.f), 6.f) * (1.f / 6.f);
  jointn[idx] = hs;
  atomicAdd(&ctx[bo], hs * (1.f / LL));
}

// ---------------------------------------------------- sigmoid gates (f and t)
__global__ void k_gates(const float* __restrict__ jointn,
                        const float* __restrict__ cf_w, const float* __restrict__ cf_b,
                        const float* __restrict__ ct_w, const float* __restrict__ ct_b,
                        float* __restrict__ sigf, float* __restrict__ sigt) {
  int idx = blockIdx.x * 256 + threadIdx.x;   // exact: B*E*F + B*E*T
  if (idx < BB * EE * FD) {
    int b = idx / (EE * FD);
    int e = (idx / FD) % EE;
    int f = idx % FD;
    float acc = cf_b[e];
    const float* w = cf_w + e * CDIM;
    const float* p = jointn + ((size_t)b * CDIM) * LL + f;
    for (int c = 0; c < CDIM; ++c) acc += w[c] * p[(size_t)c * LL];
    sigf[idx] = 1.f / (1.f + __expf(-acc));
  } else {
    int j = idx - BB * EE * FD;
    int b = j / (EE * TD);
    int e = (j / TD) % EE;
    int t = j % TD;
    float acc = ct_b[e];
    const float* w = ct_w + e * CDIM;
    const float* p = jointn + ((size_t)b * CDIM) * LL + FD + t;
    for (int c = 0; c < CDIM; ++c) acc += w[c] * p[(size_t)c * LL];
    sigt[j] = 1.f / (1.f + __expf(-acc));
  }
}

// ----------------- dynamic weights: softmax attn, mix bases, f16 weight store
__global__ void k_dyn(const float* __restrict__ ctx,
                      const float* exp_attn_w, const float* exp_attn_b, const float* exp_basis,
                      const float* dw_attn_w,  const float* dw_attn_b,  const float* dw_basis,
                      const float* proj_attn_w,const float* proj_attn_b,const float* proj_basis,
                      const float* act_w, const float* act_b,
                      const float* lambdas, const float* init_v,
                      _Float16* __restrict__ expw_h, _Float16* __restrict__ projw_h,
                      float* __restrict__ dww, float* __restrict__ rawA) {
  __shared__ float at[3][KKN];
  int b = blockIdx.y;
  int tid = threadIdx.x;
  if (tid < 3) {
    const float* W  = tid == 0 ? exp_attn_w : (tid == 1 ? dw_attn_w : proj_attn_w);
    const float* Bi = tid == 0 ? exp_attn_b : (tid == 1 ? dw_attn_b : proj_attn_b);
    float lg[KKN]; float mx = -1e30f;
    for (int k = 0; k < KKN; ++k) {
      float a = Bi[k];
      for (int c = 0; c < CDIM; ++c) a += ctx[b * CDIM + c] * W[k * CDIM + c];
      lg[k] = a; mx = fmaxf(mx, a);
    }
    float ssum = 0.f;
    for (int k = 0; k < KKN; ++k) { lg[k] = __expf(lg[k] - mx); ssum += lg[k]; }
    for (int k = 0; k < KKN; ++k) at[tid][k] = lg[k] / ssum;
  }
  __syncthreads();
  int idx = blockIdx.x * 256 + tid;           // exact: 36096 per batch
  if (idx < EE * CC) {                        // exp_w -> (E,C) f16
    float a = 0.f;
    for (int k = 0; k < KKN; ++k) a += at[0][k] * exp_basis[k * (EE * CC) + idx];
    expw_h[(size_t)b * EE * CC + idx] = (_Float16)a;
  } else if (idx < 2 * EE * CC) {             // proj_w -> (O,E) f16
    int j = idx - EE * CC;
    float a = 0.f;
    for (int k = 0; k < KKN; ++k) a += at[2][k] * proj_basis[k * (EE * OO) + j];
    projw_h[(size_t)b * EE * OO + j] = (_Float16)a;
  } else if (idx < 2 * EE * CC + EE * 9) {    // depthwise 3x3 weights
    int j = idx - 2 * EE * CC;
    float a = 0.f;
    for (int k = 0; k < KKN; ++k) a += at[1][k] * dw_basis[k * (EE * 9) + j];
    dww[(size_t)b * EE * 9 + j] = a;
  } else if (idx < 2 * EE * CC + EE * 9 + 4 * EE) {   // activation coefs
    int j = idx - (2 * EE * CC + EE * 9);
    float a = act_b[j];
    for (int c = 0; c < CDIM; ++c) a += ctx[b * CDIM + c] * act_w[j * CDIM + c];
    int jj = j / EE;
    rawA[(size_t)b * 4 * EE + j] = lambdas[jj] * a + init_v[jj];
  }
}

// -------------------------- WMMA expansion: out1 = exp_w(E x 64) @ x(64 x FT)
__global__ void __launch_bounds__(256)
k_expand(const float* __restrict__ x, const _Float16* __restrict__ expw_h,
         _Float16* __restrict__ out1h, float* sum1, float* sumsq1) {
  __shared__ float s1l[EE], sq1l[EE];
  int tid = threadIdx.x;
  s1l[tid] = 0.f; sq1l[tid] = 0.f;
  __syncthreads();
  int b     = blockIdx.y;
  int wave  = tid >> 5, lane = tid & 31;
  int mhalf = wave & 1;
  int ntile = blockIdx.x * 4 + (wave >> 1);   // grid.x = 500 -> 2000 tiles
  int col   = ntile * 16 + (lane & 15);
  int kb    = (lane >= 16) ? 8 : 0;
  int rowoff = lane & 15;

  // B fragments (x column, f32 -> f16), K = 64 in two K=32 chunks
  const float* xb = x + (size_t)b * CC * FT;
  v16h bfrag[2];
  for (int kk = 0; kk < 2; ++kk)
    for (int i = 0; i < 16; ++i) {
      int k = kk * 32 + kb + (i < 8 ? i : i + 8);
      bfrag[kk][i] = (_Float16)xb[(size_t)k * FT + col];
    }

  v8f acc[8];
  const _Float16* ab = expw_h + (size_t)b * EE * CC;
  for (int mt = 0; mt < 8; ++mt) {
    const _Float16* ap = ab + (size_t)(mhalf * 128 + mt * 16 + rowoff) * CC;
    v8f c = {};
    for (int kk = 0; kk < 2; ++kk) {
      v8h lo = *(const v8h*)(ap + kk * 32 + kb);
      v8h hi = *(const v8h*)(ap + kk * 32 + kb + 16);
      v16h a;
      for (int i = 0; i < 8; ++i) { a[i] = lo[i]; a[i + 8] = hi[i]; }
      c = __builtin_amdgcn_wmma_f32_16x16x32_f16(false, a, false, bfrag[kk],
                                                 (short)0, c, false, false);
    }
    acc[mt] = c;
  }
  for (int mt = 0; mt < 8; ++mt) {
    int m0 = mhalf * 128 + mt * 16;
    for (int r = 0; r < 8; ++r) {
      int e = m0 + r + ((lane >= 16) ? 8 : 0);
      float v = acc[mt][r];
      out1h[((size_t)(b * EE + e)) * FT + col] = (_Float16)v;
      atomicAdd(&s1l[e], v);
      atomicAdd(&sq1l[e], v * v);
    }
  }
  __syncthreads();
  atomicAdd(&sum1[tid], s1l[tid]);
  atomicAdd(&sumsq1[tid], sq1l[tid]);
}

// ---------- BN1 + gates + depthwise 3x3 (padded plane in LDS) + BN2 stats
__global__ void k_dwconv(const _Float16* __restrict__ out1h,
                         const float* sum1, const float* sumsq1,
                         const float* en_g, const float* en_b,
                         const float* __restrict__ sigf, const float* __restrict__ sigt,
                         const float* __restrict__ dww,
                         _Float16* __restrict__ out2h, float* sum2, float* sumsq2) {
  extern __shared__ float tile[];             // (FD+2) x (TD+2) fp32
  const int TW = TD + 2;
  int be = blockIdx.x;                        // b*E + e
  int e  = be & (EE - 1);
  int tid = threadIdx.x;
  float inv = 1.f / ((float)BB * FT);
  float m = sum1[e] * inv;
  float var = sumsq1[e] * inv - m * m;
  float r = en_g[e] * rsqrtf(var + EPSF);
  float s1 = r, t1 = en_b[e] - m * r;

  for (int i = tid; i < (FD + 2) * TW; i += 256) tile[i] = 0.f;
  __syncthreads();
  const _Float16* src = out1h + (size_t)be * FT;
  const float* sf = sigf + (size_t)be * FD;
  const float* st = sigt + (size_t)be * TD;
  for (int i = tid; i < FT; i += 256) {
    int f = i / TD, t = i - f * TD;
    float v = (float)src[i] * s1 + t1;
    tile[(f + 1) * TW + (t + 1)] = v * sf[f] * st[t];
  }
  __syncthreads();
  float w[9];
  for (int j = 0; j < 9; ++j) w[j] = dww[(size_t)be * 9 + j];
  float lsum = 0.f, lsq = 0.f;
  _Float16* dst = out2h + (size_t)be * FT;
  for (int i = tid; i < FT; i += 256) {
    int f = i / TD, t = i - f * TD;
    float a = 0.f;
    for (int df = 0; df < 3; ++df)
      for (int dt = 0; dt < 3; ++dt)
        a += w[df * 3 + dt] * tile[(f + df) * TW + (t + dt)];
    dst[i] = (_Float16)a;
    lsum += a; lsq += a * a;
  }
  __shared__ float red[2];
  if (tid == 0) { red[0] = 0.f; red[1] = 0.f; }
  __syncthreads();
  atomicAdd(&red[0], lsum);
  atomicAdd(&red[1], lsq);
  __syncthreads();
  if (tid == 0) { atomicAdd(&sum2[e], red[0]); atomicAdd(&sumsq2[e], red[1]); }
}

// ------------- fold BN2 into the dyn-activation: ACT4 = (A1,B1,A2,B2) per b,e
__global__ void k_actc(const float* sum2, const float* sumsq2,
                       const float* dn_g, const float* dn_b,
                       const float* __restrict__ rawA, float4* __restrict__ ACT4) {
  int idx = blockIdx.x * 256 + threadIdx.x;   // exact: B*E = 4096
  int b = idx / EE, e = idx % EE;
  float inv = 1.f / ((float)BB * FT);
  float m = sum2[e] * inv;
  float var = sumsq2[e] * inv - m * m;
  float r = dn_g[e] * rsqrtf(var + EPSF);
  float s2 = r, t2 = dn_b[e] - m * r;
  const float* ra = rawA + (size_t)b * 4 * EE;
  float a1 = ra[e], b1 = ra[EE + e], a2 = ra[2 * EE + e], b2 = ra[3 * EE + e];
  ACT4[idx] = make_float4(a1 * s2, a1 * t2 + b1, a2 * s2, a2 * t2 + b2);
}

// ---- WMMA projection: out3 = proj_w(64 x 256) @ act(bn2(out2)) (256 x FT)
__global__ void __launch_bounds__(256)
k_proj(const _Float16* __restrict__ out2h, const _Float16* __restrict__ projw_h,
       const float4* __restrict__ ACT4,
       float* __restrict__ out3, float* sum3, float* sumsq3) {
  __shared__ float s3l[OO], sq3l[OO];
  int tid = threadIdx.x;
  if (tid < OO) { s3l[tid] = 0.f; sq3l[tid] = 0.f; }
  __syncthreads();
  int b    = blockIdx.y;
  int wave = tid >> 5, lane = tid & 31;
  int ntile = blockIdx.x * 8 + wave;          // grid.x = 250 -> 2000 tiles
  int col   = ntile * 16 + (lane & 15);
  int kb    = (lane >= 16) ? 8 : 0;
  int rowoff = lane & 15;

  v8f acc[4] = {};
  const _Float16* xb = out2h + (size_t)b * EE * FT;
  const float4*  A4 = ACT4 + (size_t)b * EE;
  const _Float16* ab = projw_h + (size_t)b * OO * EE;
  for (int kk = 0; kk < 8; ++kk) {            // K = 256 in K=32 chunks
    v16h bf;
    for (int i = 0; i < 16; ++i) {
      int eidx = kk * 32 + kb + (i < 8 ? i : i + 8);
      float v = (float)xb[(size_t)eidx * FT + col];
      float4 c = A4[eidx];                    // broadcast across lanes
      float rr = fmaxf(fmaf(c.x, v, c.y), fmaf(c.z, v, c.w));
      bf[i] = (_Float16)rr;
    }
    for (int mt = 0; mt < 4; ++mt) {
      const _Float16* ap = ab + (size_t)(mt * 16 + rowoff) * EE + kk * 32 + kb;
      v8h lo = *(const v8h*)ap;
      v8h hi = *(const v8h*)(ap + 16);
      v16h a;
      for (int i = 0; i < 8; ++i) { a[i] = lo[i]; a[i + 8] = hi[i]; }
      acc[mt] = __builtin_amdgcn_wmma_f32_16x16x32_f16(false, a, false, bf,
                                                       (short)0, acc[mt], false, false);
    }
  }
  for (int mt = 0; mt < 4; ++mt)
    for (int r = 0; r < 8; ++r) {
      int o = mt * 16 + r + ((lane >= 16) ? 8 : 0);
      float v = acc[mt][r];
      out3[((size_t)(b * OO + o)) * FT + col] = v;
      atomicAdd(&s3l[o], v);
      atomicAdd(&sq3l[o], v * v);
    }
  __syncthreads();
  if (tid < OO) { atomicAdd(&sum3[tid], s3l[tid]); atomicAdd(&sumsq3[tid], sq3l[tid]); }
}

// ----------------------------------------------- BN3 + residual -> d_out
__global__ void k_final(const float* __restrict__ out3, const float* __restrict__ x,
                        const float* sum3, const float* sumsq3,
                        const float* pn_g, const float* pn_b,
                        float* __restrict__ out) {
  int bo = blockIdx.x;                        // b*O + o  (O == C, residual aligns)
  int o  = bo & (OO - 1);
  float inv = 1.f / ((float)BB * FT);
  float m = sum3[o] * inv;
  float var = sumsq3[o] * inv - m * m;
  float r = pn_g[o] * rsqrtf(var + EPSF);
  float s = r, t = pn_b[o] - m * r;
  const float* p3 = out3 + (size_t)bo * FT;
  const float* px = x + (size_t)bo * FT;
  float* po = out + (size_t)bo * FT;
  for (int i = threadIdx.x; i < FT; i += 256)
    po[i] = p3[i] * s + t + px[i];
}

extern "C" void kernel_launch(void* const* d_in, const int* in_sizes, int n_in,
                              void* d_out, int out_size, void* d_ws, size_t ws_size,
                              hipStream_t stream) {
  (void)in_sizes; (void)n_in; (void)out_size; (void)ws_size;
  const float* x          = (const float*)d_in[0];
  const float* jc_w       = (const float*)d_in[1];
  const float* jn_g       = (const float*)d_in[2];
  const float* jn_b       = (const float*)d_in[3];
  const float* cf_w       = (const float*)d_in[4];
  const float* cf_b       = (const float*)d_in[5];
  const float* ct_w       = (const float*)d_in[6];
  const float* ct_b       = (const float*)d_in[7];
  const float* exp_basis  = (const float*)d_in[8];
  const float* exp_attn_w = (const float*)d_in[9];
  const float* exp_attn_b = (const float*)d_in[10];
  const float* en_g       = (const float*)d_in[11];
  const float* en_b       = (const float*)d_in[12];
  const float* dw_basis   = (const float*)d_in[13];
  const float* dw_attn_w  = (const float*)d_in[14];
  const float* dw_attn_b  = (const float*)d_in[15];
  const float* dn_g       = (const float*)d_in[16];
  const float* dn_b       = (const float*)d_in[17];
  const float* act_w      = (const float*)d_in[18];
  const float* act_b      = (const float*)d_in[19];
  const float* lambdas    = (const float*)d_in[20];
  const float* init_v     = (const float*)d_in[21];
  const float* proj_basis = (const float*)d_in[22];
  const float* proj_attn_w= (const float*)d_in[23];
  const float* proj_attn_b= (const float*)d_in[24];
  const float* pn_g       = (const float*)d_in[25];
  const float* pn_b       = (const float*)d_in[26];

  char* ws = (char*)d_ws;
  auto alloc = [&](size_t bytes) {
    char* p = ws; ws += (bytes + 255) & ~(size_t)255; return p;
  };
  // stats (1280 f) + ctx (1024 f) contiguous -> one memset
  float* stats  = (float*)alloc(1280 * 4);
  float* ctx    = (float*)alloc((size_t)BB * CDIM * 4);
  float* jsum   = stats,        *jsumsq = stats + 64;
  float* sum1   = stats + 128,  *sumsq1 = stats + 384;
  float* sum2   = stats + 640,  *sumsq2 = stats + 896;
  float* sum3   = stats + 1152, *sumsq3 = stats + 1216;
  float* xf     = (float*)alloc((size_t)BB * CC * FD * 4);
  float* xt     = (float*)alloc((size_t)BB * CC * TD * 4);
  float* joint  = (float*)alloc((size_t)BB * CDIM * LL * 4);
  float* jointn = (float*)alloc((size_t)BB * CDIM * LL * 4);
  float* sigf   = (float*)alloc((size_t)BB * EE * FD * 4);
  float* sigt   = (float*)alloc((size_t)BB * EE * TD * 4);
  float* dww    = (float*)alloc((size_t)BB * EE * 9 * 4);
  float* rawA   = (float*)alloc((size_t)BB * 4 * EE * 4);
  float4* ACT4  = (float4*)alloc((size_t)BB * EE * 16);
  _Float16* expw_h  = (_Float16*)alloc((size_t)BB * EE * CC * 2);
  _Float16* projw_h = (_Float16*)alloc((size_t)BB * OO * EE * 2);
  _Float16* out1h   = (_Float16*)alloc((size_t)BB * EE * FT * 2);
  _Float16* out2h   = (_Float16*)alloc((size_t)BB * EE * FT * 2);
  float*    out3    = (float*)alloc((size_t)BB * OO * FT * 4);

  hipMemsetAsync(stats, 0, (1280 + BB * CDIM) * 4, stream);

  k_pool <<<dim3(BB * CC), 256, 0, stream>>>(x, xf, xt);
  k_joint<<<dim3((BB * CDIM * LL) / 256), 256, 0, stream>>>(xf, xt, jc_w, joint, jsum, jsumsq);
  k_ctx  <<<dim3((BB * CDIM * LL) / 256), 256, 0, stream>>>(joint, jsum, jsumsq, jn_g, jn_b, jointn, ctx);
  k_gates<<<dim3((BB * EE * FD + BB * EE * TD) / 256), 256, 0, stream>>>(
      jointn, cf_w, cf_b, ct_w, ct_b, sigf, sigt);
  k_dyn  <<<dim3((2 * EE * CC + EE * 9 + 4 * EE) / 256, BB), 256, 0, stream>>>(
      ctx, exp_attn_w, exp_attn_b, exp_basis, dw_attn_w, dw_attn_b, dw_basis,
      proj_attn_w, proj_attn_b, proj_basis, act_w, act_b, lambdas, init_v,
      expw_h, projw_h, dww, rawA);
  k_expand<<<dim3(FT / 16 / 4, BB), 256, 0, stream>>>(x, expw_h, out1h, sum1, sumsq1);
  size_t convLds = (size_t)(FD + 2) * (TD + 2) * 4;
  k_dwconv<<<dim3(BB * EE), 256, convLds, stream>>>(
      out1h, sum1, sumsq1, en_g, en_b, sigf, sigt, dww, out2h, sum2, sumsq2);
  k_actc <<<dim3((BB * EE) / 256), 256, 0, stream>>>(sum2, sumsq2, dn_g, dn_b, rawA, ACT4);
  k_proj <<<dim3(FT / 16 / 8, BB), 256, 0, stream>>>(out2h, projw_h, ACT4, out3, sum3, sumsq3);
  k_final<<<dim3(BB * OO), 256, 0, stream>>>(out3, x, sum3, sumsq3, pn_g, pn_b, (float*)d_out);
}
